// EEGMambaBackbone_79577154060899
// MI455X (gfx1250) — matrix operand order repaired and checked by reading, not compiled, not run on hardware
//
#include <hip/hip_runtime.h>
#include <math.h>

// ---------------- problem constants ----------------
#define Bn   32
#define CHn  19
#define Pn   30
#define Sn   200
#define DMn  200
#define NLn  12
#define DIn  400
#define DSn  64
#define DRn  13
#define Ln   (CHn*Pn)        // 570
#define Tn   (Bn*Ln)         // 18240
#define KP_DM 224            // 200 padded to mult of 32
#define KP_DI 416            // 400 padded to mult of 32

typedef __attribute__((ext_vector_type(16))) __bf16 v16bf;
typedef __attribute__((ext_vector_type(8)))  __bf16 v8bf;
typedef __attribute__((ext_vector_type(8)))  float  v8f;

// ---------------- workspace layout (float units) ----------------
static constexpr size_t O_HID = 0;
static constexpr size_t O_RES = O_HID + (size_t)Tn*DMn;
static constexpr size_t O_HNB = O_RES + (size_t)Tn*DMn;              // bf16 T*KP_DM
static constexpr size_t O_XZ  = O_HNB + (size_t)Tn*KP_DM/2;          // f32 T*800 (reused as t_raw)
static constexpr size_t O_XC  = O_XZ  + (size_t)Tn*800;
static constexpr size_t O_XCB = O_XC  + (size_t)Tn*DIn;              // bf16 T*KP_DI
static constexpr size_t O_XDB = O_XCB + (size_t)Tn*KP_DI/2;          // f32 T*141
static constexpr size_t O_DT  = O_XDB + (size_t)Tn*141;
static constexpr size_t O_YGB = O_DT  + (size_t)Tn*DIn;              // bf16 T*KP_DI
static constexpr size_t NWB   = (size_t)NLn*(800*KP_DM + 141*KP_DI + DMn*KP_DI);
static constexpr size_t O_WB  = O_YGB + (size_t)Tn*KP_DI/2;
static constexpr size_t O_DFT = O_WB  + (NWB+1)/2;
static constexpr size_t O_GNS = O_DFT + 2*101*200;
static constexpr size_t WS_FLOATS = O_GNS + 512;

static constexpr size_t WB_IN = 0;
static constexpr size_t WB_XP = WB_IN + (size_t)NLn*800*KP_DM;
static constexpr size_t WB_OP = WB_XP + (size_t)NLn*141*KP_DI;

// ---------------- small helpers ----------------
__device__ __forceinline__ float siluf(float x){ return x/(1.f+__expf(-x)); }

// f32 -> bf16 with K padding: dst[rows][Kp], src[rows][K]
__global__ void k_cvt_pad(const float* __restrict__ s, __bf16* __restrict__ d,
                          int rows, int K, int Kp){
  int i = blockIdx.x*blockDim.x + threadIdx.x;
  if (i >= rows*Kp) return;
  int n = i / Kp, k = i % Kp;
  d[i] = (k < K) ? (__bf16)s[(size_t)n*K + k] : (__bf16)0.f;
}

// zero pad columns [from, stride) of a Tn x stride bf16 buffer
__global__ void k_padzero(__bf16* __restrict__ p, int stride, int from){
  int cols = stride - from;
  int i = blockIdx.x*blockDim.x + threadIdx.x;
  if (i >= Tn*cols) return;
  int rw = i / cols, c = from + i % cols;
  p[(size_t)rw*stride + c] = (__bf16)0.f;
}

// DFT tables: cos/sin(2*pi*f*s/200), exact integer modulo for accuracy
__global__ void k_dft_init(float* __restrict__ ct, float* __restrict__ st){
  int i = blockIdx.x*blockDim.x + threadIdx.x;
  if (i >= 101*200) return;
  int f = i/200, s = i%200;
  int m = (f*s) % 200;
  float w = 6.28318530717958647692f * (float)m * (1.f/200.f);
  ct[i] = cosf(w); st[i] = sinf(w);
}

// patch-embedding time conv: t[b,oc,l,w] = sum_k x[b,l, w*25-24+k] * pw[oc,k]
__global__ void k_timeconv(const float* __restrict__ x, const float* __restrict__ pw,
                           float* __restrict__ traw){
  int i = blockIdx.x*blockDim.x + threadIdx.x;
  if (i >= Bn*25*Ln*8) return;
  int w  = i & 7;
  int l  = (i>>3) % Ln;
  int oc = (i/(8*Ln)) % 25;
  int b  = i/(8*Ln*25);
  const float* xr = x + ((size_t)b*Ln + l)*Sn;
  const float* kr = pw + oc*49;
  int s0 = w*25 - 24;
  float acc = 0.f;
  #pragma unroll
  for (int k=0;k<49;++k){ int s=s0+k; if (s>=0 && s<Sn) acc += xr[s]*kr[k]; }
  traw[i] = acc;
}

// GroupNorm stats: per (b, group of 5 channels) over 5*570*8 elems
__global__ void k_gnstats(const float* __restrict__ traw, float* __restrict__ gns){
  int bg = blockIdx.x; int b = bg/5, g = bg%5;
  float s=0.f, ss=0.f;
  for (int e=threadIdx.x; e<5*Ln*8; e+=blockDim.x){
    int sub = e/(Ln*8), rem = e%(Ln*8);
    float v = traw[(size_t)(b*25 + g*5 + sub)*(Ln*8) + rem];
    s += v; ss += v*v;
  }
  __shared__ float a1[256], a2[256];
  a1[threadIdx.x]=s; a2[threadIdx.x]=ss; __syncthreads();
  for (int st=128; st; st>>=1){
    if (threadIdx.x<st){ a1[threadIdx.x]+=a1[threadIdx.x+st]; a2[threadIdx.x]+=a2[threadIdx.x+st]; }
    __syncthreads();
  }
  if (threadIdx.x==0){
    float inv = 1.f/(5.f*Ln*8.f);
    float m = a1[0]*inv;
    gns[bg*2] = m; gns[bg*2+1] = a2[0]*inv - m*m;
  }
}

// per-token: spectral DFT magnitude x spec_w  +  GN+GELU(time conv)  -> pe
__global__ void k_pe(const float* __restrict__ x, const float* __restrict__ traw,
                     const float* __restrict__ gns, const float* __restrict__ gnw,
                     const float* __restrict__ gnb, const float* __restrict__ specw,
                     const float* __restrict__ ct, const float* __restrict__ st,
                     float* __restrict__ pe){
  int t = blockIdx.x; int b = t/Ln; int l = t%Ln;
  __shared__ float bins[101];
  const float* xr = x + (size_t)t*Sn;
  for (int f=threadIdx.x; f<101; f+=blockDim.x){
    float re=0.f, im=0.f;
    const float* c = ct + f*200; const float* sn = st + f*200;
    for (int s=0;s<Sn;++s){ float v=xr[s]; re += v*c[s]; im += v*sn[s]; }
    bins[f] = sqrtf(re*re + im*im) * (1.f/200.f);
  }
  __syncthreads();
  for (int d=threadIdx.x; d<DMn; d+=blockDim.x){
    const float* swr = specw + d*101;
    float spec = 0.f;
    for (int f=0; f<101; ++f) spec += bins[f]*swr[f];
    int oc = d>>3, w = d&7;
    float v = traw[((size_t)(b*25+oc)*Ln + l)*8 + w];
    int g = oc/5;
    float m  = gns[(b*5+g)*2], va = gns[(b*5+g)*2+1];
    float nv = (v-m)*rsqrtf(va+1e-5f)*gnw[oc] + gnb[oc];
    float ge = 0.5f*nv*(1.f + erff(nv*0.70710678118654752440f));
    pe[(size_t)t*DMn + d] = ge + spec;
  }
}

// depthwise 7x7 positional conv over (ch,pn), hidden = pe + pos
__global__ void k_posconv(const float* __restrict__ pe, const float* __restrict__ pw,
                          float* __restrict__ hid){
  int i = blockIdx.x*blockDim.x + threadIdx.x;
  if (i >= Tn*DMn) return;
  int d = i % DMn; int t = i / DMn;
  int p = t % Pn; int c = (t/Pn) % CHn; int b = t / Ln;
  float acc = pe[i];
  const float* kr = pw + d*49;
  #pragma unroll
  for (int di=0; di<7; ++di){
    int c2 = c + di - 3; if ((unsigned)c2 >= CHn) continue;
    #pragma unroll
    for (int dj=0; dj<7; ++dj){
      int p2 = p + dj - 3; if ((unsigned)p2 >= Pn) continue;
      acc += pe[((size_t)(b*CHn + c2)*Pn + p2)*DMn + d] * kr[di*7+dj];
    }
  }
  hid[i] = acc;
}

// residual accumulate + LayerNorm -> bf16 activations (row stride KP_DM, pad zeroed)
__global__ void k_ln(const float* __restrict__ hid, float* __restrict__ res,
                     const float* __restrict__ w, const float* __restrict__ bi,
                     __bf16* __restrict__ hnb, int first){
  int t = blockIdx.x; int d = threadIdx.x;
  float v = 0.f;
  if (d < DMn){ v = hid[(size_t)t*DMn + d]; if (!first) v += res[(size_t)t*DMn + d]; }
  __shared__ float a1[256], a2[256];
  a1[d] = (d<DMn)?v:0.f; a2[d] = (d<DMn)?v*v:0.f; __syncthreads();
  for (int st=128; st; st>>=1){
    if (d<st){ a1[d]+=a1[d+st]; a2[d]+=a2[d+st]; } __syncthreads();
  }
  float m = a1[0]*(1.f/DMn);
  float var = a2[0]*(1.f/DMn) - m*m;
  if (d < DMn){
    res[(size_t)t*DMn + d] = v;
    float nv = (v-m)*rsqrtf(var+1e-5f)*w[d] + bi[d];
    hnb[(size_t)t*KP_DM + d] = (__bf16)nv;
  } else if (d < KP_DM){
    hnb[(size_t)t*KP_DM + d] = (__bf16)0.f;
  }
}

// ------------- WMMA bf16 GEMM:  Y[Tn,N] = X[Tn,KP] * W[N,KP]^T (K pre-padded) ---
// KP is compile-time: the K loop fully unrolls -> branch-free load/wmma stream.
// 4 waves/block; block tile 32(M) x 64(N); each wave: 2 accumulators (32x16),
// B fragment reused across both; A tile double-buffered in LDS; all loads 16B+.
template<int KP>
__global__ void __launch_bounds__(128)
k_gemm_bf16(const __bf16* __restrict__ X, const __bf16* __restrict__ W,
            float* __restrict__ Y, int N){
  constexpr int NK = KP >> 5;
  __shared__ __bf16 As[2][32][40];          // stride 40 keeps 16B alignment
  const int tid = threadIdx.x;
  const int lane = tid & 31, wv = tid >> 5;
  const int m0 = blockIdx.x * 32;
  const int n  = blockIdx.y * 64 + wv*16 + (lane & 15);
  const int hlf = lane >> 4;
  const int row = lane & 15;
  const __bf16* wr = W + (size_t)((n < N) ? n : (N-1)) * KP;
  const int lr = (tid*8) >> 5;              // cooperative A-load: row 0..31
  const int lc = (tid*8) & 31;              // col 0,8,16,24
  const __bf16* xsrc = X + (size_t)(m0+lr)*KP + lc;

  v8f acc0 = {0.f,0.f,0.f,0.f,0.f,0.f,0.f,0.f};
  v8f acc1 = {0.f,0.f,0.f,0.f,0.f,0.f,0.f,0.f};

  // prologue: tile 0 -> buffer 0
  *(v8bf*)&As[0][lr][lc] = *(const v8bf*)(xsrc);

  #pragma unroll
  for (int t = 0; t < NK; ++t){
    const int c = t & 1;                    // compile-time after unroll
    __syncthreads();                        // buf c ready; buf c^1 free
    if (t+1 < NK)                           // folds statically
      *(v8bf*)&As[c^1][lr][lc] = *(const v8bf*)(xsrc + (size_t)(t+1)*32);
    const int k0 = t*32;
    // A fragments (ISA 16-bit 16x32 lane swizzle), two M tiles
    v8bf alo0 = *(const v8bf*)&As[c][row     ][hlf*8];
    v8bf ahi0 = *(const v8bf*)&As[c][row     ][16 + hlf*8];
    v8bf alo1 = *(const v8bf*)&As[c][row + 16][hlf*8];
    v8bf ahi1 = *(const v8bf*)&As[c][row + 16][16 + hlf*8];
    v16bf a0 = __builtin_shufflevector(alo0, ahi0, 0,1,2,3,4,5,6,7,8,9,10,11,12,13,14,15);
    v16bf a1 = __builtin_shufflevector(alo1, ahi1, 0,1,2,3,4,5,6,7,8,9,10,11,12,13,14,15);
    // B fragment: same swizzle as A (K=k0+hlf*8..+7, then k0+16+hlf*8..+7)
    v8bf blo = *(const v8bf*)(wr + k0 + hlf*8);
    v8bf bhi = *(const v8bf*)(wr + k0 + 16 + hlf*8);
    v16bf bf = __builtin_shufflevector(blo, bhi, 0,1,2,3,4,5,6,7,8,9,10,11,12,13,14,15);
    acc0 = __builtin_amdgcn_wmma_f32_16x16x32_bf16(false, a0, false, bf,
                                                   (short)0, acc0, false, false);
    acc1 = __builtin_amdgcn_wmma_f32_16x16x32_bf16(false, a1, false, bf,
                                                   (short)0, acc1, false, false);
  }
  if (n < N){
    #pragma unroll
    for (int r=0; r<8; ++r){
      int m = m0 + r + hlf*8;
      Y[(size_t)m*N + n]        = acc0[r];
      Y[(size_t)(m+16)*N + n]   = acc1[r];
    }
  }
}

// causal depthwise conv1d (k=4) + SiLU; emits f32 (scan) and bf16 (x_proj, stride KP_DI)
__global__ void k_conv1d(const float* __restrict__ xz, const float* __restrict__ cw,
                         const float* __restrict__ cb, float* __restrict__ xc,
                         __bf16* __restrict__ xcb){
  int i = blockIdx.x*blockDim.x + threadIdx.x;
  if (i >= Tn*DIn) return;
  int d = i % DIn; int t = i / DIn;
  int l = t % Ln;  int b = t / Ln;
  const float* k4 = cw + d*4;
  float acc = cb[d];
  #pragma unroll
  for (int j=0;j<4;++j){
    int l2 = l - 3 + j; if (l2 < 0) continue;
    acc += xz[(size_t)(b*Ln + l2)*800 + d] * k4[j];
  }
  float sv = siluf(acc);
  xc[i] = sv;
  xcb[(size_t)t*KP_DI + d] = (__bf16)sv;
}

// dt = softplus(dtr @ dtw^T + dtb)   (K=13 -> VALU)
__global__ void k_dt(const float* __restrict__ xdbl, const float* __restrict__ dtw,
                     const float* __restrict__ dtb, float* __restrict__ dt){
  int i = blockIdx.x*blockDim.x + threadIdx.x;
  if (i >= Tn*DIn) return;
  int d = i % DIn; int t = i / DIn;
  const float* xr = xdbl + (size_t)t*141;
  const float* wr = dtw + d*DRn;
  float acc = dtb[d];
  #pragma unroll
  for (int r=0;r<DRn;++r) acc += xr[r]*wr[r];
  dt[i] = (acc > 20.f) ? acc : log1pf(__expf(acc));
}

// selective scan: one wave32 per (b,d); 2 states/lane; shfl_xor reduce; fused SiLU gate
__global__ void k_scan(const float* __restrict__ xc, const float* __restrict__ dt,
                       const float* __restrict__ xdbl, const float* __restrict__ xz,
                       const float* __restrict__ Alog, const float* __restrict__ Dsk,
                       __bf16* __restrict__ ygb){
  int tid = threadIdx.x; int lane = tid & 31; int wv = tid >> 5;
  int g = blockIdx.x*8 + wv;          // 0..12799
  int b = g / DIn, d = g % DIn;
  int n0 = lane*2;
  float A0 = -__expf(Alog[(size_t)d*DSn + n0]);
  float A1 = -__expf(Alog[(size_t)d*DSn + n0 + 1]);
  float Dp = Dsk[d];
  float h0 = 0.f, h1 = 0.f;
  for (int l=0; l<Ln; ++l){
    size_t base = (size_t)(b*Ln + l);
    float dtt = dt[base*DIn + d];
    float u   = xc[base*DIn + d];
    const float* xr = xdbl + base*141;
    float Bv0 = xr[DRn + n0],        Bv1 = xr[DRn + n0 + 1];
    float Cv0 = xr[DRn + DSn + n0],  Cv1 = xr[DRn + DSn + n0 + 1];
    float du = dtt*u;
    h0 = h0*__expf(dtt*A0) + du*Bv0;
    h1 = h1*__expf(dtt*A1) + du*Bv1;
    float part = h0*Cv0 + h1*Cv1;
    #pragma unroll
    for (int off=16; off; off>>=1) part += __shfl_xor(part, off, 32);
    if (lane == 0){
      float y = part + u*Dp;
      float z = xz[base*800 + DIn + d];
      ygb[base*KP_DI + d] = (__bf16)(y * siluf(z));
    }
  }
}

// final residual + LayerNorm -> f32 output
__global__ void k_final(const float* __restrict__ hid, const float* __restrict__ res,
                        const float* __restrict__ w, const float* __restrict__ bi,
                        float* __restrict__ out){
  int t = blockIdx.x; int d = threadIdx.x;
  float v = 0.f;
  if (d < DMn) v = hid[(size_t)t*DMn + d] + res[(size_t)t*DMn + d];
  __shared__ float a1[256], a2[256];
  a1[d] = (d<DMn)?v:0.f; a2[d] = (d<DMn)?v*v:0.f; __syncthreads();
  for (int st=128; st; st>>=1){
    if (d<st){ a1[d]+=a1[d+st]; a2[d]+=a2[d+st]; } __syncthreads();
  }
  float m = a1[0]*(1.f/DMn);
  float var = a2[0]*(1.f/DMn) - m*m;
  if (d < DMn)
    out[(size_t)t*DMn + d] = (v-m)*rsqrtf(var+1e-5f)*w[d] + bi[d];
}

// ---------------- host orchestration ----------------
extern "C" void kernel_launch(void* const* d_in, const int* in_sizes, int n_in,
                              void* d_out, int out_size, void* d_ws, size_t ws_size,
                              hipStream_t stream){
  if (ws_size < WS_FLOATS*sizeof(float)) return;
  const float* x    = (const float*)d_in[0];
  const float* pecw = (const float*)d_in[1];
  const float* gnw  = (const float*)d_in[2];
  const float* gnb  = (const float*)d_in[3];
  const float* spw  = (const float*)d_in[4];
  const float* posw = (const float*)d_in[5];
  const float* ipw  = (const float*)d_in[6];
  const float* c1w  = (const float*)d_in[7];
  const float* c1b  = (const float*)d_in[8];
  const float* xpw  = (const float*)d_in[9];
  const float* dtw  = (const float*)d_in[10];
  const float* dtb  = (const float*)d_in[11];
  const float* alog = (const float*)d_in[12];
  const float* dsk  = (const float*)d_in[13];
  const float* opw  = (const float*)d_in[14];
  const float* lnw  = (const float*)d_in[15];
  const float* lnb  = (const float*)d_in[16];
  const float* nfw  = (const float*)d_in[17];
  const float* nfb  = (const float*)d_in[18];
  float* ws = (float*)d_ws;

  float*  hid  = ws + O_HID;
  float*  res  = ws + O_RES;      // also holds pe in the front end
  __bf16* hnb  = (__bf16*)(ws + O_HNB);
  float*  xz   = ws + O_XZ;
  float*  traw = ws + O_XZ;       // reuse (front end finishes before layers)
  float*  xc   = ws + O_XC;
  __bf16* xcb  = (__bf16*)(ws + O_XCB);
  float*  xdbl = ws + O_XDB;
  float*  dtv  = ws + O_DT;
  __bf16* ygb  = (__bf16*)(ws + O_YGB);
  __bf16* wb   = (__bf16*)(ws + O_WB);
  float*  dftc = ws + O_DFT;
  float*  dfts = ws + O_DFT + 101*200;
  float*  gns  = ws + O_GNS;

  const int TB = 256;
  auto gr = [](int n, int b){ return (n + b - 1)/b; };

  // weight conversion + K padding (deterministic, rerun every call)
  { int n = NLn*800*KP_DM; k_cvt_pad<<<gr(n,TB),TB,0,stream>>>(ipw, wb+WB_IN, NLn*800, DMn, KP_DM); }
  { int n = NLn*141*KP_DI; k_cvt_pad<<<gr(n,TB),TB,0,stream>>>(xpw, wb+WB_XP, NLn*141, DIn, KP_DI); }
  { int n = NLn*DMn*KP_DI; k_cvt_pad<<<gr(n,TB),TB,0,stream>>>(opw, wb+WB_OP, NLn*DMn, DIn, KP_DI); }
  // zero activation K-pads once per call
  k_padzero<<<gr(Tn*(KP_DI-DIn),TB),TB,0,stream>>>(xcb, KP_DI, DIn);
  k_padzero<<<gr(Tn*(KP_DI-DIn),TB),TB,0,stream>>>(ygb, KP_DI, DIn);
  k_dft_init<<<gr(101*200,TB),TB,0,stream>>>(dftc, dfts);

  // front end
  k_timeconv<<<gr(Bn*25*Ln*8,TB),TB,0,stream>>>(x, pecw, traw);
  k_gnstats<<<Bn*5, TB, 0, stream>>>(traw, gns);
  k_pe<<<Tn, TB, 0, stream>>>(x, traw, gns, gnw, gnb, spw, dftc, dfts, res);
  k_posconv<<<gr(Tn*DMn,TB),TB,0,stream>>>(res, posw, hid);

  // Mamba layers
  for (int i=0; i<NLn; ++i){
    k_ln<<<Tn, TB, 0, stream>>>(hid, res, lnw+i*DMn, lnb+i*DMn, hnb, (i==0)?1:0);
    { dim3 g(Tn/32, (800+63)/64);
      k_gemm_bf16<KP_DM><<<g, 128, 0, stream>>>(hnb, wb+WB_IN+(size_t)i*800*KP_DM, xz, 800); }
    k_conv1d<<<gr(Tn*DIn,TB),TB,0,stream>>>(xz, c1w+i*DIn*4, c1b+i*DIn, xc, xcb);
    { dim3 g(Tn/32, (141+63)/64);
      k_gemm_bf16<KP_DI><<<g, 128, 0, stream>>>(xcb, wb+WB_XP+(size_t)i*141*KP_DI, xdbl, 141); }
    k_dt<<<gr(Tn*DIn,TB),TB,0,stream>>>(xdbl, dtw+i*DIn*DRn, dtb+i*DIn, dtv);
    k_scan<<<(Bn*DIn)/8, TB, 0, stream>>>(xc, dtv, xdbl, xz,
                                          alog+(size_t)i*DIn*DSn, dsk+i*DIn, ygb);
    { dim3 g(Tn/32, (DMn+63)/64);
      k_gemm_bf16<KP_DI><<<g, 128, 0, stream>>>(ygb, wb+WB_OP+(size_t)i*DMn*KP_DI, hid, DMn); }
  }

  k_final<<<Tn, TB, 0, stream>>>(hid, res, nfw, nfb, (float*)d_out);
}
